// SelfAttentionBlock_31980326486088
// MI455X (gfx1250) — matrix operand order repaired and checked by reading, not compiled, not run on hardware
//
#include <hip/hip_runtime.h>

#define BATCH 4
#define CIN   256
#define NPIX  4096   // 64*64
#define LAT   256
#define LDK   272    // padded LDS row stride (elements): 544 B = 17*32 B

typedef __attribute__((ext_vector_type(16))) __bf16 v16bf;
typedef __attribute__((ext_vector_type(8)))  __bf16 v8bf;
typedef __attribute__((ext_vector_type(8)))  float  v8f;

// ---------------------------------------------------------------------------
// WMMA helper (CDNA5 wave32, V_WMMA_F32_16X16X32_BF16)
// ---------------------------------------------------------------------------
static __device__ inline v8f wmma_bf16(v16bf a, v16bf b, v8f c) {
    return __builtin_amdgcn_wmma_f32_16x16x32_bf16(
        /*neg_a=*/false, a, /*neg_b=*/false, b,
        /*c_mod=*/(short)0, c, /*reuse_a=*/false, /*reuse_b=*/false);
}

// A fragment: 16x32 (MxK), row-major source, leading dim ld (elements).
// Lane l holds row (l&15); K = 16*(l>>4) .. +15.
static __device__ inline v16bf load_afrag(const __bf16* __restrict__ base, int ld, int lane) {
    int row = lane & 15;
    int hi  = lane >> 4;
    const __bf16* p = base + row * ld + hi * 16;
    v8bf lo  = *(const v8bf*)(p);
    v8bf hi8 = *(const v8bf*)(p + 8);
    v16bf r;
#pragma unroll
    for (int i = 0; i < 8; ++i) { r[i] = lo[i]; r[i + 8] = hi8[i]; }
    return r;
}

// B fragment: 32x16 (KxN) from the TRANSPOSED matrix (N rows x K cols),
// row-major, leading dim ld. Lane l holds column (l&15), K = 16*(l>>4) .. +15.
static __device__ inline v16bf load_bfrag(const __bf16* __restrict__ baseT, int ld, int lane) {
    int col = lane & 15;
    int hi  = lane >> 4;
    return *(const v16bf*)(baseT + col * ld + hi * 16);
}

// ---------------------------------------------------------------------------
// CDNA5 async copy (GLOBAL_LOAD_ASYNC_TO_LDS_B128, tracked by ASYNCcnt).
// ldsaddr: per-lane LDS byte address; g: per-lane global source (16 B).
// ---------------------------------------------------------------------------
static __device__ inline unsigned lds_addr(const void* p) {
    // generic shared pointer = {aperture_hi, 32-bit LDS offset}
    return (unsigned)(unsigned long long)p;
}
static __device__ inline void async_ld_b128(unsigned ldsaddr, const void* g) {
    asm volatile("global_load_async_to_lds_b128 %0, %1, off"
                 :: "v"(ldsaddr), "v"((unsigned long long)g) : "memory");
}
static __device__ inline void wait_async0() {
    asm volatile("s_wait_asynccnt 0x0" ::: "memory");
}

// ---------------------------------------------------------------------------
// Kernel 0a: x [B][C][N] fp32  ->  xT [B][N][C] bf16
// ---------------------------------------------------------------------------
__global__ void cvt_x_kernel(const float* __restrict__ x, __bf16* __restrict__ xT) {
    int idx = blockIdx.x * blockDim.x + threadIdx.x;     // over B*CIN*NPIX
    int n = idx & (NPIX - 1);
    int c = (idx >> 12) & (CIN - 1);
    int b = idx >> 20;
    xT[((size_t)b * NPIX + n) * CIN + c] = (__bf16)x[idx];
}

// ---------------------------------------------------------------------------
// Kernel 0b: weights fp32 -> bf16 (Wq/Wk/Wv [L][C], Wo [C][L])
// ---------------------------------------------------------------------------
__global__ void cvt_w_kernel(const float* __restrict__ Wq, const float* __restrict__ Wk,
                             const float* __restrict__ Wv, const float* __restrict__ Wo,
                             __bf16* __restrict__ Wqb, __bf16* __restrict__ Wkb,
                             __bf16* __restrict__ Wvb, __bf16* __restrict__ Wob) {
    int idx = blockIdx.x * blockDim.x + threadIdx.x;     // over 256*256
    Wqb[idx] = (__bf16)Wq[idx];
    Wkb[idx] = (__bf16)Wk[idx];
    Wvb[idx] = (__bf16)Wv[idx];
    Wob[idx] = (__bf16)Wo[idx];
}

// ---------------------------------------------------------------------------
// Kernel 1: fused Q/K projection. qT/kT[b][n][l] = xT[b][n][:]·W[l][:] + bias
// ---------------------------------------------------------------------------
__global__ __launch_bounds__(128) void qk_proj_kernel(
        const __bf16* __restrict__ xT,
        const __bf16* __restrict__ Wqb, const __bf16* __restrict__ Wkb,
        const float* __restrict__ bq, const float* __restrict__ bk,
        __bf16* __restrict__ qT, __bf16* __restrict__ kT) {
    int wave = (blockIdx.x * blockDim.x + threadIdx.x) >> 5;
    int lane = threadIdx.x & 31;
    int b  = wave >> 12;
    int nt = (wave >> 4) & 255;
    int lt = wave & 15;

    const __bf16* xbase = xT + ((size_t)b * NPIX + nt * 16) * CIN;
    v8f accq = {}; v8f acck = {};
#pragma unroll
    for (int kc = 0; kc < 8; ++kc) {
        v16bf a  = load_afrag(xbase + kc * 32, CIN, lane);
        v16bf fq = load_bfrag(Wqb + lt * 16 * CIN + kc * 32, CIN, lane);
        v16bf fk = load_bfrag(Wkb + lt * 16 * CIN + kc * 32, CIN, lane);
        accq = wmma_bf16(a, fq, accq);
        acck = wmma_bf16(a, fk, acck);
    }
    int l  = lt * 16 + (lane & 15);
    int hi = lane >> 4;
    float biasq = bq[l], biask = bk[l];
#pragma unroll
    for (int r = 0; r < 8; ++r) {
        int n = nt * 16 + r + 8 * hi;
        size_t off = ((size_t)b * NPIX + n) * LAT + l;
        qT[off] = (__bf16)(accq[r] + biasq);
        kT[off] = (__bf16)(acck[r] + biask);
    }
}

// ---------------------------------------------------------------------------
// Kernel 2: V projection.  v[b][l][n] = Wv[l][:]·xT[b][n][:] + bv[l]
// ---------------------------------------------------------------------------
__global__ __launch_bounds__(128) void v_proj_kernel(
        const __bf16* __restrict__ xT, const __bf16* __restrict__ Wvb,
        const float* __restrict__ bv, __bf16* __restrict__ vM) {
    int wave = (blockIdx.x * blockDim.x + threadIdx.x) >> 5;
    int lane = threadIdx.x & 31;
    int b  = wave >> 12;
    int lt = (wave >> 8) & 15;
    int nt = wave & 255;

    v8f acc = {};
#pragma unroll
    for (int kc = 0; kc < 8; ++kc) {
        v16bf a = load_afrag(Wvb + lt * 16 * CIN + kc * 32, CIN, lane);
        v16bf f = load_bfrag(xT + ((size_t)b * NPIX + nt * 16) * CIN + kc * 32, CIN, lane);
        acc = wmma_bf16(a, f, acc);
    }
    int n  = nt * 16 + (lane & 15);
    int hi = lane >> 4;
#pragma unroll
    for (int r = 0; r < 8; ++r) {
        int l = lt * 16 + r + 8 * hi;
        vM[((size_t)b * LAT + l) * NPIX + n] = (__bf16)(acc[r] + bv[l]);
    }
}

// ---------------------------------------------------------------------------
// Kernel 3: fused flash attention (softmax over i, per column j).
// 4 waves/block; each wave owns a 16-column j-tile (64 columns per block).
// kT tiles + qT i-blocks staged in LDS via GLOBAL_LOAD_ASYNC_TO_LDS_B128;
// the next qT stage overlaps with the O-phase WMMAs. Rows padded to 544 B
// to rotate LDS banks. Output: resT[b][j][c] bf16.
// ---------------------------------------------------------------------------
__global__ __launch_bounds__(128) void attn_kernel(
        const __bf16* __restrict__ qT, const __bf16* __restrict__ kT,
        const __bf16* __restrict__ vM, __bf16* __restrict__ resT) {
    __shared__ __attribute__((aligned(64))) __bf16 lds_kt[4 * 16 * LDK]; // 34 KB
    __shared__ __attribute__((aligned(64))) __bf16 lds_qt[32 * LDK];     // 17 KB
    __shared__ __attribute__((aligned(64))) __bf16 lds_pt[4 * 16 * 32];  //  4 KB

    int tid  = threadIdx.x;
    int lane = tid & 31;
    int w    = tid >> 5;                       // wave id 0..3
    int b    = blockIdx.x >> 6;                // 64 blocks per batch
    int jt4  = blockIdx.x & 63;                // 64-column group

    const __bf16* qTb = qT + (size_t)b * NPIX * LAT;
    const __bf16* vb  = vM + (size_t)b * LAT * NPIX;
    __bf16* myKt = lds_kt + w * 16 * LDK;
    __bf16* myPt = lds_pt + w * 16 * 32;

    // ---- stage this wave's kT tile (16 rows x 512 B) via async copy ----
    {
        const __bf16* ksrc = kT + ((size_t)b * NPIX + jt4 * 64 + w * 16) * LAT;
        unsigned dst0 = lds_addr(myKt);
#pragma unroll
        for (int t = 0; t < 16; ++t)
            async_ld_b128(dst0 + t * (LDK * 2) + lane * 16, ksrc + t * LAT + lane * 8);
    }
    // ---- stage first qT i-block (32 rows x 512 B, cooperative) ----
    {
        unsigned dst0 = lds_addr(lds_qt);
#pragma unroll
        for (int t = 0; t < 8; ++t) {
            int chunk = tid + 128 * t;         // 0..1023
            int row = chunk >> 5, col = chunk & 31;
            async_ld_b128(dst0 + row * (LDK * 2) + col * 16, qTb + row * LAT + col * 8);
        }
    }
    wait_async0();
    __syncthreads();

    v8f acc[16];
#pragma unroll
    for (int ct = 0; ct < 16; ++ct)
#pragma unroll
        for (int r = 0; r < 8; ++r) acc[ct][r] = 0.f;

    float m = -3.0e38f, lsum = 0.f;
    const float scale = 0.0625f;               // 1/sqrt(256)

    for (int i0 = 0; i0 < NPIX; i0 += 32) {
        // ---- S[i, j] for 32 i-rows x 16 j-cols (operands from LDS) ----
        v8f s0 = {}; v8f s1 = {};
#pragma unroll
        for (int kc = 0; kc < 8; ++kc) {
            v16bf fk = load_bfrag(myKt + kc * 32, LDK, lane);
            v16bf a0 = load_afrag(lds_qt + kc * 32, LDK, lane);
            v16bf a1 = load_afrag(lds_qt + 16 * LDK + kc * 32, LDK, lane);
            s0 = wmma_bf16(a0, fk, s0);
            s1 = wmma_bf16(a1, fk, s1);
        }
        // ---- online softmax over i (lanes l and l+16 share a column) ----
        float tm = -3.0e38f;
#pragma unroll
        for (int r = 0; r < 8; ++r) {
            s0[r] *= scale; s1[r] *= scale;
            tm = fmaxf(tm, fmaxf(s0[r], s1[r]));
        }
        tm = fmaxf(tm, __shfl_xor(tm, 16, 32));
        float mnew = fmaxf(m, tm);
        float corr = __expf(m - mnew);
        float rsum = 0.f;
        v8f p0, p1;
#pragma unroll
        for (int r = 0; r < 8; ++r) {
            p0[r] = __expf(s0[r] - mnew);
            p1[r] = __expf(s1[r] - mnew);
            rsum += p0[r] + p1[r];
        }
        rsum += __shfl_xor(rsum, 16, 32);
        lsum = lsum * corr + rsum;
        m = mnew;
#pragma unroll
        for (int ct = 0; ct < 16; ++ct)
#pragma unroll
            for (int r = 0; r < 8; ++r) acc[ct][r] *= corr;

        // ---- P^T -> LDS ([16 j][32 i] bf16, per-wave region) ----
        {
            int j = lane & 15, hi = lane >> 4;
            v8bf pk0, pk1;
#pragma unroll
            for (int r = 0; r < 8; ++r) { pk0[r] = (__bf16)p0[r]; pk1[r] = (__bf16)p1[r]; }
            *(v8bf*)(myPt + j * 32 + 8 * hi)      = pk0;  // i = r + 8*hi
            *(v8bf*)(myPt + j * 32 + 16 + 8 * hi) = pk1;  // i = 16 + r + 8*hi
        }
        __syncthreads();   // P visible; all waves done reading lds_qt
        v16bf pf = *(const v16bf*)(myPt + (lane & 15) * 32 + (lane >> 4) * 16);

        // ---- overlap: async-stage next qT i-block while O-phase runs ----
        int inext = i0 + 32;
        if (inext < NPIX) {
            unsigned dst0 = lds_addr(lds_qt);
            const __bf16* src = qTb + (size_t)inext * LAT;
#pragma unroll
            for (int t = 0; t < 8; ++t) {
                int chunk = tid + 128 * t;
                int row = chunk >> 5, col = chunk & 31;
                async_ld_b128(dst0 + row * (LDK * 2) + col * 16, src + row * LAT + col * 8);
            }
        }

        // ---- O[c, j] += V[c, i-block] * P  (V streams from L2) ----
#pragma unroll
        for (int ct = 0; ct < 16; ++ct) {
            v16bf av = load_afrag(vb + (size_t)(ct * 16) * NPIX + i0, NPIX, lane);
            acc[ct] = wmma_bf16(av, pf, acc[ct]);
        }

        wait_async0();
        __syncthreads();   // qT buffer refilled; lds_pt free for rewrite
    }

    // ---- normalize and store resT[b][j][c] (16-byte packed stores) ----
    float inv = 1.f / lsum;
    int j  = jt4 * 64 + w * 16 + (lane & 15);
    int hi = lane >> 4;
#pragma unroll
    for (int ct = 0; ct < 16; ++ct) {
        v8bf outp;
#pragma unroll
        for (int r = 0; r < 8; ++r) outp[r] = (__bf16)(acc[ct][r] * inv);
        *(v8bf*)(resT + ((size_t)b * NPIX + j) * LAT + ct * 16 + 8 * hi) = outp;
    }
}

// ---------------------------------------------------------------------------
// Kernel 4: output projection. out[b][c][n] = Wo[c][:]·resT[b][n][:] + bo[c]
// ---------------------------------------------------------------------------
__global__ __launch_bounds__(128) void out_proj_kernel(
        const __bf16* __restrict__ Wob, const __bf16* __restrict__ resT,
        const float* __restrict__ bo, float* __restrict__ out) {
    int wave = (blockIdx.x * blockDim.x + threadIdx.x) >> 5;
    int lane = threadIdx.x & 31;
    int b  = wave >> 12;
    int ct = (wave >> 8) & 15;
    int nt = wave & 255;

    v8f acc = {};
#pragma unroll
    for (int kc = 0; kc < 8; ++kc) {
        v16bf a = load_afrag(Wob + ct * 16 * LAT + kc * 32, LAT, lane);
        v16bf f = load_bfrag(resT + ((size_t)b * NPIX + nt * 16) * LAT + kc * 32, LAT, lane);
        acc = wmma_bf16(a, f, acc);
    }
    int n  = nt * 16 + (lane & 15);
    int hi = lane >> 4;
#pragma unroll
    for (int r = 0; r < 8; ++r) {
        int c = ct * 16 + r + 8 * hi;
        out[((size_t)b * CIN + c) * NPIX + n] = acc[r] + bo[c];
    }
}

// ---------------------------------------------------------------------------
// Launch
// ---------------------------------------------------------------------------
extern "C" void kernel_launch(void* const* d_in, const int* in_sizes, int n_in,
                              void* d_out, int out_size, void* d_ws, size_t ws_size,
                              hipStream_t stream) {
    const float* x  = (const float*)d_in[0];
    const float* Wq = (const float*)d_in[1];
    const float* bq = (const float*)d_in[2];
    const float* Wk = (const float*)d_in[3];
    const float* bk = (const float*)d_in[4];
    const float* Wv = (const float*)d_in[5];
    const float* bv = (const float*)d_in[6];
    const float* Wo = (const float*)d_in[7];
    const float* bo = (const float*)d_in[8];
    float* out = (float*)d_out;

    char* ws = (char*)d_ws;
    const size_t SZ_BNL = (size_t)BATCH * NPIX * LAT * sizeof(__bf16);  // 8 MB each
    __bf16* xT   = (__bf16*)(ws);
    __bf16* qT   = (__bf16*)(ws + 1 * SZ_BNL);
    __bf16* kT   = (__bf16*)(ws + 2 * SZ_BNL);
    __bf16* vM   = (__bf16*)(ws + 3 * SZ_BNL);
    __bf16* resT = (__bf16*)(ws + 4 * SZ_BNL);
    char*   wbase = ws + 5 * SZ_BNL;
    const size_t SZ_W = (size_t)LAT * CIN * sizeof(__bf16);             // 128 KB
    __bf16* Wqb = (__bf16*)(wbase);
    __bf16* Wkb = (__bf16*)(wbase + 1 * SZ_W);
    __bf16* Wvb = (__bf16*)(wbase + 2 * SZ_W);
    __bf16* Wob = (__bf16*)(wbase + 3 * SZ_W);

    cvt_x_kernel<<<(BATCH * CIN * NPIX) / 256, 256, 0, stream>>>(x, xT);
    cvt_w_kernel<<<(LAT * CIN) / 256, 256, 0, stream>>>(Wq, Wk, Wv, Wo, Wqb, Wkb, Wvb, Wob);
    qk_proj_kernel<<<4096, 128, 0, stream>>>(xT, Wqb, Wkb, bq, bk, qT, kT);
    v_proj_kernel<<<4096, 128, 0, stream>>>(xT, Wvb, bv, vM);
    // 4 waves/block, 64 j-columns per block: 256 blocks
    attn_kernel<<<BATCH * (NPIX / 64), 128, 0, stream>>>(qT, kT, vM, resT);
    out_proj_kernel<<<4096, 128, 0, stream>>>(Wob, resT, bo, out);
}
